// FPModule_20160576487543
// MI455X (gfx1250) — matrix operand order repaired and checked by reading, not compiled, not run on hardware
//
#include <hip/hip_runtime.h>
#include <hip/hip_bf16.h>

typedef __attribute__((ext_vector_type(16))) _Float16 v16h;
typedef __attribute__((ext_vector_type(8)))  float    v8f;

// Fixed problem sizes from setup_inputs()
constexpr int B_  = 8;
constexpr int N_  = 8192;
constexpr int M_  = 2048;
constexpr int C_  = 128;   // feature channels
constexpr int CIN = 256;   // 2*C concat
constexpr int H1  = 256;
constexpr int H2  = 128;
constexpr int S_  = B_ * N_;   // 65536 flattened samples

// ---------------------------------------------------------------------------
// Kernel 1: 3-NN search.  One thread per query point, xyz2 tiled through LDS.
// ---------------------------------------------------------------------------
__global__ void three_nn_kernel(const float* __restrict__ xyz1,
                                const float* __restrict__ xyz2,
                                int* __restrict__ idx3,
                                float* __restrict__ w3) {
    __shared__ float sx[256 * 3];
    const int tid = threadIdx.x;
    const int b   = blockIdx.x / (N_ / 256);
    const int n   = (blockIdx.x % (N_ / 256)) * 256 + tid;
    const int s   = b * N_ + n;

    const float px = xyz1[s * 3 + 0];
    const float py = xyz1[s * 3 + 1];
    const float pz = xyz1[s * 3 + 2];

    float d0 = 3e38f, d1 = 3e38f, d2 = 3e38f;
    int   i0 = 0,     i1 = 0,     i2 = 0;

    for (int m0 = 0; m0 < M_; m0 += 256) {
        __syncthreads();
        const int mm = m0 + tid;
        sx[tid * 3 + 0] = xyz2[(b * M_ + mm) * 3 + 0];
        sx[tid * 3 + 1] = xyz2[(b * M_ + mm) * 3 + 1];
        sx[tid * 3 + 2] = xyz2[(b * M_ + mm) * 3 + 2];
        __syncthreads();
        for (int j = 0; j < 256; ++j) {
            const float dx = px - sx[j * 3 + 0];
            const float dy = py - sx[j * 3 + 1];
            const float dz = pz - sx[j * 3 + 2];
            const float d  = dx * dx + dy * dy + dz * dz;
            const int   mi = m0 + j;
            if (d < d2) {
                if (d < d1) {
                    if (d < d0) { d2 = d1; i2 = i1; d1 = d0; i1 = i0; d0 = d; i0 = mi; }
                    else        { d2 = d1; i2 = i1; d1 = d;  i1 = mi; }
                } else          { d2 = d;  i2 = mi; }
            }
        }
    }
    const float r0 = 1.0f / (d0 + 1e-8f);
    const float r1 = 1.0f / (d1 + 1e-8f);
    const float r2 = 1.0f / (d2 + 1e-8f);
    const float rs = 1.0f / (r0 + r1 + r2);
    idx3[s * 3 + 0] = i0;  w3[s * 3 + 0] = r0 * rs;
    idx3[s * 3 + 1] = i1;  w3[s * 3 + 1] = r1 * rs;
    idx3[s * 3 + 2] = i2;  w3[s * 3 + 2] = r2 * rs;
}

// ---------------------------------------------------------------------------
// Kernel 2: f32 -> f16 cast (for weight matrices)
// ---------------------------------------------------------------------------
__global__ void cast_f16_kernel(const float* __restrict__ in,
                                _Float16* __restrict__ out, int n) {
    const int i = blockIdx.x * 256 + threadIdx.x;
    if (i < n) out[i] = (_Float16)in[i];
}

// ---------------------------------------------------------------------------
// Kernel 3: build concat input X (s-major: X[s*256 + c]) in f16.
// c<128 : 3-point interpolation of features2; c>=128 : copy features1.
// ---------------------------------------------------------------------------
__global__ void build_x_kernel(const float* __restrict__ f1,
                               const float* __restrict__ f2,
                               const int* __restrict__ idx3,
                               const float* __restrict__ w3,
                               _Float16* __restrict__ Xh) {
    const int i = blockIdx.x * 256 + threadIdx.x;   // 0 .. S*256-1
    const int s = i >> 8;
    const int c = i & 255;
    const int b = s >> 13;      // /8192
    const int n = s & 8191;
    float v;
    if (c < C_) {
        const float* base = f2 + (size_t)(b * C_ + c) * M_;
        const int j0 = idx3[s * 3 + 0];
        const int j1 = idx3[s * 3 + 1];
        const int j2 = idx3[s * 3 + 2];
        v = w3[s * 3 + 0] * base[j0] + w3[s * 3 + 1] * base[j1] +
            w3[s * 3 + 2] * base[j2];
    } else {
        v = f1[(size_t)(b * C_ + (c - C_)) * N_ + n];
    }
    Xh[(size_t)s * CIN + c] = (_Float16)v;
}

// ---------------------------------------------------------------------------
// Kernel 4: WMMA f16 GEMM, software-pipelined (double-buffered fragments).
// Y[s, o] = sum_k W[o, k] * X[s, k] + bias[o]
// W: o-major (k contiguous).  X: s-major (k contiguous).  Y: s-major f16.
// Each wave computes a 64(o) x 16(s) strip: B-fragment reused across 4 WMMAs.
// Templated so the k-loop fully unrolls and the double-buffer index is static.
// sched_barrier(0) fences keep next-step loads ABOVE current WMMAs so the
// register allocator cannot collapse the two fragment generations.
// ---------------------------------------------------------------------------
union FragU { v16h v; unsigned u[8]; };

template <int MOUT, int KD>
__global__ void gemm_wmma_f16(const _Float16* __restrict__ Wh,
                              const _Float16* __restrict__ Xh,
                              const float* __restrict__ bias,
                              _Float16* __restrict__ Yh, int Stot) {
    const int wave  = blockIdx.x * (blockDim.x >> 5) + (threadIdx.x >> 5);
    const int lane  = threadIdx.x & 31;
    constexpr int tiles = MOUT >> 6;             // 64-wide o strips
    const int obase = (wave % tiles) << 6;
    const int s0    = (wave / tiles) << 4;
    if (s0 >= Stot) return;

    const int mn   = lane & 15;
    const int half = lane >> 4;

    const _Float16* brow = Xh + (size_t)(s0 + mn) * KD;

    // B fragment (32x16): element pair p holds K = (p<4?0:16) + half*8 + 2(p&3) + w
    // (same dword mapping as A).  A fragment (16x32): element pair p
    // holds K = (p<4?0:16) + half*8 + 2(p&3) + w -> dwords
    // (p<4?0:8) + half*4 + (p&3)  (two contiguous runs of 4 dwords).
    auto loadB = [&](int k0, FragU& f) {
        const unsigned* bp = (const unsigned*)(brow + k0);
        #pragma unroll
        for (int p = 0; p < 8; ++p) f.u[p] = bp[((p < 4) ? 0 : 8) + half * 4 + (p & 3)];
    };
    auto loadA = [&](int k0, int q, FragU& f) {
        const unsigned* ap =
            (const unsigned*)(Wh + (size_t)(obase + q * 16 + mn) * KD + k0);
        #pragma unroll
        for (int p = 0; p < 8; ++p)
            f.u[p] = ap[((p < 4) ? 0 : 8) + half * 4 + (p & 3)];
    };

    FragU fa[2][4];
    FragU fb[2];
    v8f   acc[4] = {};

    loadB(0, fb[0]);
    #pragma unroll
    for (int q = 0; q < 4; ++q) loadA(0, q, fa[0][q]);

    constexpr int KSTEPS = KD / 32;
    #pragma unroll
    for (int kk = 0; kk < KSTEPS; ++kk) {
        const int cur = kk & 1;
        const int nxt = cur ^ 1;
        if (kk + 1 < KSTEPS) {
            loadB((kk + 1) * 32, fb[nxt]);
            #pragma unroll
            for (int q = 0; q < 4; ++q) loadA((kk + 1) * 32, q, fa[nxt][q]);
        }
        // Phase fence: next-step loads stay above this point; the WMMAs below
        // may not be reordered with them -> true double buffering.
        __builtin_amdgcn_sched_barrier(0);
        #pragma unroll
        for (int q = 0; q < 4; ++q) {
            acc[q] = __builtin_amdgcn_wmma_f32_16x16x32_f16(
                false, fa[cur][q].v, false, fb[cur].v, (short)0, acc[q],
                false, false);
        }
        __builtin_amdgcn_sched_barrier(0);
    }

    #pragma unroll
    for (int q = 0; q < 4; ++q) {
        const int o0 = obase + q * 16;
        #pragma unroll
        for (int r = 0; r < 8; ++r) {
            const int m = r + 8 * half;                 // D layout: m = r + 8*half
            const float v = acc[q][r] + bias[o0 + m];
            Yh[(size_t)(s0 + mn) * MOUT + o0 + m] = (_Float16)v;
        }
    }
}

// ---------------------------------------------------------------------------
// Kernel 5: per-channel batch statistics (biased var, like torch/jnp BN train)
// ---------------------------------------------------------------------------
__global__ void bn_stats_kernel(const _Float16* __restrict__ Yh,
                                float* __restrict__ meanOut,
                                float* __restrict__ rstdOut,
                                int Cc, int Stot) {
    __shared__ float ss[256], sq[256];
    const int c = blockIdx.x;
    const int t = threadIdx.x;
    float s = 0.f, q = 0.f;
    for (int i = t; i < Stot; i += 256) {
        if (i + 256 < Stot)
            __builtin_prefetch(&Yh[(size_t)(i + 256) * Cc + c], 0, 1);
        const float v = (float)Yh[(size_t)i * Cc + c];
        s += v; q += v * v;
    }
    ss[t] = s; sq[t] = q;
    __syncthreads();
    for (int w = 128; w > 0; w >>= 1) {
        if (t < w) { ss[t] += ss[t + w]; sq[t] += sq[t + w]; }
        __syncthreads();
    }
    if (t == 0) {
        const float mu  = ss[0] / (float)Stot;
        const float var = sq[0] / (float)Stot - mu * mu;
        meanOut[c] = mu;
        rstdOut[c] = rsqrtf(var + 1e-5f);
    }
}

// ---------------------------------------------------------------------------
// Kernel 6: BN + ReLU in place (f16, s-major), 2 elements per thread (dword).
// ---------------------------------------------------------------------------
__global__ void bn_apply_f16_kernel(_Float16* __restrict__ Yh,
                                    const float* __restrict__ mean,
                                    const float* __restrict__ rstd,
                                    const float* __restrict__ g,
                                    const float* __restrict__ be,
                                    int Cc) {
    const int i  = blockIdx.x * 256 + threadIdx.x;   // pair index
    const int e0 = i * 2;
    const int c0 = e0 & (Cc - 1);
    const int c1 = c0 + 1;                           // Cc is even; same sample s
    float v0 = (float)Yh[e0];
    float v1 = (float)Yh[e0 + 1];
    v0 = (v0 - mean[c0]) * rstd[c0] * g[c0] + be[c0];
    v1 = (v1 - mean[c1]) * rstd[c1] * g[c1] + be[c1];
    Yh[e0]     = (_Float16)(v0 > 0.f ? v0 : 0.f);
    Yh[e0 + 1] = (_Float16)(v1 > 0.f ? v1 : 0.f);
}

// ---------------------------------------------------------------------------
// Kernel 7: BN + ReLU, f32 output in (B, H2, N) layout.
// Thread index == flat output index -> fully coalesced stores; the strided
// f16 reads (256B stride) hit the L2-resident Y2 buffer.
// ---------------------------------------------------------------------------
__global__ void bn_apply_out_kernel(const _Float16* __restrict__ Yh,
                                    const float* __restrict__ mean,
                                    const float* __restrict__ rstd,
                                    const float* __restrict__ g,
                                    const float* __restrict__ be,
                                    float* __restrict__ out) {
    const int i   = blockIdx.x * 256 + threadIdx.x;  // flat (b, c, n) index
    const int b   = i >> 20;                         // / (128*8192)
    const int rem = i & ((1 << 20) - 1);
    const int c   = rem >> 13;
    const int n   = rem & 8191;
    const int s   = b * N_ + n;
    float v = (float)Yh[(size_t)s * H2 + c];
    v = (v - mean[c]) * rstd[c] * g[c] + be[c];
    out[i] = v > 0.f ? v : 0.f;
}

// ---------------------------------------------------------------------------
extern "C" void kernel_launch(void* const* d_in, const int* in_sizes, int n_in,
                              void* d_out, int out_size, void* d_ws, size_t ws_size,
                              hipStream_t stream) {
    const float* xyz1 = (const float*)d_in[0];
    const float* xyz2 = (const float*)d_in[1];
    const float* f1   = (const float*)d_in[2];
    const float* f2   = (const float*)d_in[3];
    const float* W1   = (const float*)d_in[4];
    const float* b1   = (const float*)d_in[5];
    const float* g1   = (const float*)d_in[6];
    const float* be1  = (const float*)d_in[7];
    const float* W2   = (const float*)d_in[8];
    const float* b2   = (const float*)d_in[9];
    const float* g2   = (const float*)d_in[10];
    const float* be2  = (const float*)d_in[11];
    float* out = (float*)d_out;

    char* ws = (char*)d_ws;
    size_t off = 0;
    int*      idx3  = (int*)(ws + off);       off += (size_t)S_ * 3 * sizeof(int);
    float*    w3    = (float*)(ws + off);     off += (size_t)S_ * 3 * sizeof(float);
    _Float16* W1h   = (_Float16*)(ws + off);  off += (size_t)H1 * CIN * 2;
    _Float16* W2h   = (_Float16*)(ws + off);  off += (size_t)H2 * H1 * 2;
    float*    mean1 = (float*)(ws + off);     off += H1 * sizeof(float);
    float*    rstd1 = (float*)(ws + off);     off += H1 * sizeof(float);
    float*    mean2 = (float*)(ws + off);     off += H2 * sizeof(float);
    float*    rstd2 = (float*)(ws + off);     off += H2 * sizeof(float);
    _Float16* Xh    = (_Float16*)(ws + off);  off += (size_t)S_ * CIN * 2;
    _Float16* Y1h   = (_Float16*)(ws + off);  off += (size_t)S_ * H1 * 2;
    _Float16* Y2h   = Xh;   // X is dead after GEMM1; reuse for Y2

    // 1) 3-NN + interpolation weights
    three_nn_kernel<<<B_ * (N_ / 256), 256, 0, stream>>>(xyz1, xyz2, idx3, w3);

    // 2) weights to f16
    cast_f16_kernel<<<(H1 * CIN + 255) / 256, 256, 0, stream>>>(W1, W1h, H1 * CIN);
    cast_f16_kernel<<<(H2 * H1 + 255) / 256, 256, 0, stream>>>(W2, W2h, H2 * H1);

    // 3) build concat(interp, f1) input, f16 s-major
    build_x_kernel<<<S_, 256, 0, stream>>>(f1, f2, idx3, w3, Xh);

    // 4) MLP layer 1: (256 x 256) x (256 x 65536) via WMMA
    {
        const int waves = (H1 / 64) * (S_ / 16);   // 4 * 4096
        gemm_wmma_f16<H1, CIN><<<waves / 4, 128, 0, stream>>>(W1h, Xh, b1, Y1h, S_);
    }
    bn_stats_kernel<<<H1, 256, 0, stream>>>(Y1h, mean1, rstd1, H1, S_);
    bn_apply_f16_kernel<<<(S_ * H1) / 512, 256, 0, stream>>>(Y1h, mean1, rstd1, g1, be1, H1);

    // 5) MLP layer 2: (128 x 256) x (256 x 65536) via WMMA
    {
        const int waves = (H2 / 64) * (S_ / 16);   // 2 * 4096
        gemm_wmma_f16<H2, H1><<<waves / 4, 128, 0, stream>>>(W2h, Y1h, b2, Y2h, S_);
    }
    bn_stats_kernel<<<H2, 256, 0, stream>>>(Y2h, mean2, rstd2, H2, S_);
    bn_apply_out_kernel<<<(S_ * H2) / 256, 256, 0, stream>>>(Y2h, mean2, rstd2, g2, be2, out);

    (void)in_sizes; (void)n_in; (void)out_size; (void)ws_size;
}
